// TransformerEncoderBlock_5720896438840
// MI455X (gfx1250) — matrix operand order repaired and checked by reading, not compiled, not run on hardware
//
#include <hip/hip_runtime.h>

// ---------------- problem constants (fixed by reference) ----------------
constexpr int Bb   = 8;
constexpr int Ss   = 2048;
constexpr int Dd   = 256;
constexpr int Hh   = 8;
constexpr int HDd  = 32;   // head dim
constexpr int FFNd = 2048;
constexpr int NR   = Bb * Ss;  // 16384 rows

// ---------------- WMMA types / helpers ----------------
typedef __attribute__((ext_vector_type(16))) __bf16 v16bf;
typedef __attribute__((ext_vector_type(8)))  float  v8f;

union V16U { v16bf v; uint4 q[2]; };

// A-operand (16x32 bf16): lane holds row m = lane&15, k = 16*(lane>>4) + {0..15}
// -> two 16B loads at p and p+8 elements.
static __device__ inline v16bf load_a_frag(const __bf16* p) {
  V16U u;
  u.q[0] = *(const uint4*)(p);
  u.q[1] = *(const uint4*)(p + 8);
  return u.v;
}
// B-operand (32x16 bf16, stored as Bt[n][k]): lane holds col n = lane&15,
// k = 16*(lane>>4) + {0..15} -> one contiguous 32B load.
static __device__ inline v16bf load_b_frag(const __bf16* p) {
  V16U u;
  u.q[0] = *(const uint4*)(p);
  u.q[1] = *(const uint4*)(p + 8);
  return u.v;
}

static __device__ inline v8f wmma_bf16f32(v16bf a, v16bf b, v8f c) {
  // D = A*B + C ; fp32 accumulate
  return __builtin_amdgcn_wmma_f32_16x16x32_bf16(false, a, false, b, (short)0, c,
                                                 false, false);
}

// ---------------- DPP 16-lane-row butterfly reductions (no LDS traffic) ----------------
// ctrl: 0xB1 = quad_perm [1,0,3,2] (xor1), 0x4E = quad_perm [2,3,0,1] (xor2),
//       0x141 = row_half_mirror (mixes the two quads of an 8-group),
//       0x140 = row_mirror (mixes the two 8-groups of a 16-lane row).
template<int CTRL>
static __device__ inline float dpp_mov(float x) {
  return __int_as_float(
      __builtin_amdgcn_update_dpp(0, __float_as_int(x), CTRL, 0xF, 0xF, true));
}
static __device__ inline float rowmax16(float x) {
  x = fmaxf(x, dpp_mov<0xB1>(x));
  x = fmaxf(x, dpp_mov<0x4E>(x));
  x = fmaxf(x, dpp_mov<0x141>(x));
  x = fmaxf(x, dpp_mov<0x140>(x));
  return x;
}
static __device__ inline float rowsum16(float x) {
  x += dpp_mov<0xB1>(x);
  x += dpp_mov<0x4E>(x);
  x += dpp_mov<0x141>(x);
  x += dpp_mov<0x140>(x);
  return x;
}

// ---------------- prep: qk_in = bf16(x+pe), x_bf = bf16(x) ----------------
__global__ __launch_bounds__(256) void prep_kernel(const float* __restrict__ x,
                                                   const float* __restrict__ pe,
                                                   __bf16* __restrict__ qkin,
                                                   __bf16* __restrict__ xbf, int n) {
  int i = blockIdx.x * 256 + threadIdx.x;
  if (i < n) {
    float xv = x[i];
    qkin[i] = (__bf16)(xv + pe[i]);
    xbf[i]  = (__bf16)xv;
  }
}

// ---------------- weight transpose+convert: Wt[n][k] = bf16(W[k][n]) ----------------
__global__ __launch_bounds__(256) void transpose_bf16_kernel(const float* __restrict__ W,
                                                             __bf16* __restrict__ Wt,
                                                             int Krows, int Ncols) {
  int i = blockIdx.x * 256 + threadIdx.x;
  if (i < Krows * Ncols) {
    int k = i / Ncols, n = i % Ncols;
    Wt[(size_t)n * Krows + k] = (__bf16)W[i];
  }
}

// ---------------- generic WMMA GEMM: C = A[M,K] x Bt[N,K]^T + bias ----------------
// 64x64 C macro-tile per wave: 4 A-frags + 4 B-frags -> 16 WMMAs per 8 loads
// (~32 flops/byte from cache, 2x the 32x32 version).
// OUTMODE: 0 = fp32 row-major, 1 = bf16 row-major, 2 = bf16 head-transposed V layout
template<int OUTMODE, int RELU>
__global__ __launch_bounds__(32) void gemm_wmma_bf16(
    const __bf16* __restrict__ A, int lda,
    const __bf16* __restrict__ Bt, int ldb,
    const float* __restrict__ bias,
    float* __restrict__ Cf, __bf16* __restrict__ Cb, int ldc, int Ktot) {
  const int l  = threadIdx.x;
  const int lg = l >> 4;     // lane group (0/1)
  const int ln = l & 15;
  const int m0 = blockIdx.y * 64;
  const int n0 = blockIdx.x * 64;

  const __bf16* arow[4];
  const __bf16* brow[4];
#pragma unroll
  for (int i = 0; i < 4; i++) {
    arow[i] = A  + (size_t)(m0 + i * 16 + ln) * lda + lg * 16;
    brow[i] = Bt + (size_t)(n0 + i * 16 + ln) * ldb + lg * 16;
  }

  v8f acc[4][4] = {{{}, {}, {}, {}}, {{}, {}, {}, {}},
                   {{}, {}, {}, {}}, {{}, {}, {}, {}}};
  for (int kk = 0; kk < Ktot; kk += 32) {
    v16bf a[4];
#pragma unroll
    for (int i = 0; i < 4; i++) a[i] = load_a_frag(arow[i] + kk);
#pragma unroll
    for (int j = 0; j < 4; j++) {
      v16bf bf = load_b_frag(brow[j] + kk);
#pragma unroll
      for (int i = 0; i < 4; i++) acc[i][j] = wmma_bf16f32(a[i], bf, acc[i][j]);
    }
  }
#pragma unroll
  for (int ti = 0; ti < 4; ti++)
#pragma unroll
    for (int tj = 0; tj < 4; tj++) {
#pragma unroll
      for (int r = 0; r < 8; r++) {
        int row = m0 + ti * 16 + lg * 8 + r;   // C layout: M = 8*(lane>>4)+vgpr
        int col = n0 + tj * 16 + ln;           //           N = lane&15
        float v = acc[ti][tj][r] + bias[col];
        if (RELU) v = fmaxf(v, 0.0f);
        if (OUTMODE == 0) {
          Cf[(size_t)row * ldc + col] = v;
        } else if (OUTMODE == 1) {
          Cb[(size_t)row * ldc + col] = (__bf16)v;
        } else {  // V head-transposed: Vt[(b*H+h)*HD + hd][s]
          int bb = row >> 11;          // / Ss
          int s  = row & (Ss - 1);
          int h  = col >> 5;           // / HDd
          int hd = col & (HDd - 1);
          Cb[(((size_t)(bb * Hh + h)) * HDd + hd) * Ss + s] = (__bf16)v;
        }
      }
    }
}

// ---------------- flash attention: one wave per (b, h, 16-query tile) ----------------
// 64-key chunks: 4 score WMMAs + 4 context WMMAs, one DPP max/sum butterfly
// per output row per chunk; prefetch of the next K/V chunk each iteration.
__global__ __launch_bounds__(32) void attn_kernel(
    const __bf16* __restrict__ Q,    // [NR, Dd] bf16
    const __bf16* __restrict__ Km,   // [NR, Dd] bf16
    const __bf16* __restrict__ Vt,   // [B*H, HD, S] bf16 (head-transposed)
    const float*  __restrict__ mask, // [B, S]
    __bf16* __restrict__ Ctx,        // [NR, Dd] bf16 context out
    float scale) {
  const int l  = threadIdx.x;
  const int lg = l >> 4;
  const int ln = l & 15;
  const int b  = blockIdx.z, h = blockIdx.y;
  const int q0 = blockIdx.x * 16;

  __shared__ __align__(16) __bf16 P[16 * 64];  // prob tile, C-layout -> A-layout staging

  const __bf16* qptr = Q + ((size_t)(b * Ss + q0 + ln)) * Dd + h * HDd + lg * 16;
  v16bf Aq = load_a_frag(qptr);  // Q tile held in registers for entire key loop

  float rmax[8], rsum[8];
  v8f acc0 = {}, acc1 = {};
#pragma unroll
  for (int r = 0; r < 8; r++) { rmax[r] = -3.0e30f; rsum[r] = 0.0f; }
  const float* mrow = mask + b * Ss;

  for (int c = 0; c < Ss; c += 64) {
    // ---- scores: S = Q (16x32) x K_chunk^T -> four 16x16 tiles (64 keys)
    const __bf16* kbase = Km + ((size_t)(b * Ss + c + ln)) * Dd + h * HDd + lg * 16;
    v8f s[4];
    float mv[4];
#pragma unroll
    for (int tj = 0; tj < 4; tj++) {
      v16bf Bk = load_b_frag(kbase + (size_t)(tj * 16) * Dd);
      v8f z = {};
      s[tj] = wmma_bf16f32(Aq, Bk, z);
      mv[tj] = mrow[c + tj * 16 + ln] * -1e9f;
    }
    // prefetch next chunk's K rows (speculative; dropped if past end)
    if (c + 64 < Ss) __builtin_prefetch(kbase + (size_t)64 * Dd, 0, 1);

    // ---- online softmax (rows striped across 16-lane groups; DPP butterflies)
#pragma unroll
    for (int r = 0; r < 8; r++) {
      float x0 = s[0][r] * scale + mv[0];
      float x1 = s[1][r] * scale + mv[1];
      float x2 = s[2][r] * scale + mv[2];
      float x3 = s[3][r] * scale + mv[3];
      float t = fmaxf(fmaxf(x0, x1), fmaxf(x2, x3));
      t = rowmax16(t);
      float nm   = fmaxf(rmax[r], t);
      float corr = __expf(rmax[r] - nm);
      float p0   = __expf(x0 - nm);
      float p1   = __expf(x1 - nm);
      float p2   = __expf(x2 - nm);
      float p3   = __expf(x3 - nm);
      float ps   = rowsum16((p0 + p1) + (p2 + p3));
      rsum[r] = rsum[r] * corr + ps;
      acc0[r] *= corr;
      acc1[r] *= corr;
      rmax[r] = nm;
      int prow = lg * 8 + r;
      P[prow * 64 + ln]      = (__bf16)p0;
      P[prow * 64 + 16 + ln] = (__bf16)p1;
      P[prow * 64 + 32 + ln] = (__bf16)p2;
      P[prow * 64 + 48 + ln] = (__bf16)p3;
    }
    __syncthreads();  // single wave; lowers to waitcnt + nop

    // ---- context: acc += P (16 x 64keys) x V_chunk (64keys x 32hd)
    v16bf Pa0 = load_a_frag(&P[ln * 64 + lg * 16]);        // keys c..c+31
    v16bf Pa1 = load_a_frag(&P[ln * 64 + 32 + lg * 16]);   // keys c+32..c+63
    const __bf16* vrow0 = Vt + ((size_t)((b * Hh + h) * HDd + ln)) * Ss + c + lg * 16;
    const __bf16* vrow1 = vrow0 + (size_t)16 * Ss;        // hd half 1
    v16bf V00 = load_b_frag(vrow0);        // keys 0-31,  hd 0-15
    v16bf V10 = load_b_frag(vrow0 + 32);   // keys 32-63, hd 0-15
    v16bf V01 = load_b_frag(vrow1);        // keys 0-31,  hd 16-31
    v16bf V11 = load_b_frag(vrow1 + 32);   // keys 32-63, hd 16-31
    if (c + 64 < Ss) {
      __builtin_prefetch(vrow0 + 64, 0, 1);
      __builtin_prefetch(vrow1 + 64, 0, 1);
    }
    acc0 = wmma_bf16f32(Pa0, V00, acc0);
    acc0 = wmma_bf16f32(Pa1, V10, acc0);
    acc1 = wmma_bf16f32(Pa0, V01, acc1);
    acc1 = wmma_bf16f32(Pa1, V11, acc1);
    __syncthreads();
  }

#pragma unroll
  for (int r = 0; r < 8; r++) {
    float inv = 1.0f / rsum[r];
    int row = q0 + lg * 8 + r;
    size_t base = ((size_t)(b * Ss + row)) * Dd + h * HDd;
    Ctx[base + ln]      = (__bf16)(acc0[r] * inv);
    Ctx[base + 16 + ln] = (__bf16)(acc1[r] * inv);
  }
}

// ---------------- residual + LayerNorm (biased var, eps=1e-5) ----------------
__global__ __launch_bounds__(256) void residual_ln_kernel(
    const float* __restrict__ a, const float* __restrict__ res,
    const float* __restrict__ gamma, const float* __restrict__ beta,
    float* __restrict__ out_f, __bf16* __restrict__ out_b) {
  int row = blockIdx.x, t = threadIdx.x;
  size_t idx = (size_t)row * Dd + t;
  float v = a[idx] + res[idx];
  __shared__ float s1[256], s2[256];
  s1[t] = v;
  s2[t] = v * v;
  __syncthreads();
  for (int off = 128; off > 0; off >>= 1) {
    if (t < off) { s1[t] += s1[t + off]; s2[t] += s2[t + off]; }
    __syncthreads();
  }
  float mean = s1[0] * (1.0f / Dd);
  float var  = s2[0] * (1.0f / Dd) - mean * mean;
  float y = gamma[t] * (v - mean) * rsqrtf(var + 1e-5f) + beta[t];
  out_f[idx] = y;
  if (out_b) out_b[idx] = (__bf16)y;
}

// ---------------- host orchestration ----------------
extern "C" void kernel_launch(void* const* d_in, const int* in_sizes, int n_in,
                              void* d_out, int out_size, void* d_ws, size_t ws_size,
                              hipStream_t stream) {
  (void)in_sizes; (void)n_in; (void)out_size; (void)ws_size;
  const float* x    = (const float*)d_in[0];
  const float* pe   = (const float*)d_in[1];
  const float* mask = (const float*)d_in[2];
  const float* Wq   = (const float*)d_in[3];
  const float* bq   = (const float*)d_in[4];
  const float* Wk   = (const float*)d_in[5];
  const float* bk   = (const float*)d_in[6];
  const float* Wv   = (const float*)d_in[7];
  const float* bv   = (const float*)d_in[8];
  const float* Wo   = (const float*)d_in[9];
  const float* bo   = (const float*)d_in[10];
  const float* g1   = (const float*)d_in[11];
  const float* b1   = (const float*)d_in[12];
  const float* W1   = (const float*)d_in[13];
  const float* bw1  = (const float*)d_in[14];
  const float* W2   = (const float*)d_in[15];
  const float* bw2  = (const float*)d_in[16];
  const float* g2   = (const float*)d_in[17];
  const float* b2   = (const float*)d_in[18];
  float* out = (float*)d_out;

  char* w = (char*)d_ws;
  auto carve = [&](size_t bytes) -> char* {
    char* p = w;
    w += (bytes + 255) & ~(size_t)255;
    return p;
  };
  __bf16* xbf   = (__bf16*)carve((size_t)NR * Dd * 2);
  __bf16* qkin  = (__bf16*)carve((size_t)NR * Dd * 2);
  __bf16* qbf   = (__bf16*)carve((size_t)NR * Dd * 2);
  __bf16* kbf   = (__bf16*)carve((size_t)NR * Dd * 2);
  __bf16* vtbf  = (__bf16*)carve((size_t)NR * Dd * 2);
  __bf16* ctxbf = (__bf16*)carve((size_t)NR * Dd * 2);
  __bf16* x1bf  = (__bf16*)carve((size_t)NR * Dd * 2);
  __bf16* hbf   = (__bf16*)carve((size_t)NR * FFNd * 2);
  float*  attnf = (float*)carve((size_t)NR * Dd * 4);
  float*  x1f   = (float*)carve((size_t)NR * Dd * 4);
  float*  ffn2f = (float*)carve((size_t)NR * Dd * 4);
  __bf16* wqt   = (__bf16*)carve((size_t)Dd * Dd * 2);
  __bf16* wkt   = (__bf16*)carve((size_t)Dd * Dd * 2);
  __bf16* wvt   = (__bf16*)carve((size_t)Dd * Dd * 2);
  __bf16* wot   = (__bf16*)carve((size_t)Dd * Dd * 2);
  __bf16* w1t   = (__bf16*)carve((size_t)Dd * FFNd * 2);
  __bf16* w2t   = (__bf16*)carve((size_t)FFNd * Dd * 2);

  // 1) prep activations
  {
    int n = NR * Dd;
    prep_kernel<<<(n + 255) / 256, 256, 0, stream>>>(x, pe, qkin, xbf, n);
  }
  // 2) weight transposes (fp32 -> bf16, N-major)
  transpose_bf16_kernel<<<(Dd * Dd + 255) / 256, 256, 0, stream>>>(Wq, wqt, Dd, Dd);
  transpose_bf16_kernel<<<(Dd * Dd + 255) / 256, 256, 0, stream>>>(Wk, wkt, Dd, Dd);
  transpose_bf16_kernel<<<(Dd * Dd + 255) / 256, 256, 0, stream>>>(Wv, wvt, Dd, Dd);
  transpose_bf16_kernel<<<(Dd * Dd + 255) / 256, 256, 0, stream>>>(Wo, wot, Dd, Dd);
  transpose_bf16_kernel<<<(Dd * FFNd + 255) / 256, 256, 0, stream>>>(W1, w1t, Dd, FFNd);
  transpose_bf16_kernel<<<(FFNd * Dd + 255) / 256, 256, 0, stream>>>(W2, w2t, FFNd, Dd);

  dim3 gProj(Dd / 64, NR / 64);    // (4, 256)
  dim3 gFfn1(FFNd / 64, NR / 64);  // (32, 256)

  // 3) Q, K projections from (x+pe); V projection from x (head-transposed output)
  gemm_wmma_bf16<1, 0><<<gProj, 32, 0, stream>>>(qkin, Dd, wqt, Dd, bq, nullptr, qbf, Dd, Dd);
  gemm_wmma_bf16<1, 0><<<gProj, 32, 0, stream>>>(qkin, Dd, wkt, Dd, bk, nullptr, kbf, Dd, Dd);
  gemm_wmma_bf16<2, 0><<<gProj, 32, 0, stream>>>(xbf,  Dd, wvt, Dd, bv, nullptr, vtbf, 0, Dd);

  // 4) flash attention
  {
    dim3 gAttn(Ss / 16, Hh, Bb);  // (128, 8, 8)
    attn_kernel<<<gAttn, 32, 0, stream>>>(qbf, kbf, vtbf, mask, ctxbf,
                                          0.17677669529663687f /* 1/sqrt(32) */);
  }

  // 5) output projection (fp32) -> residual + LN1 -> x1 (fp32 + bf16)
  gemm_wmma_bf16<0, 0><<<gProj, 32, 0, stream>>>(ctxbf, Dd, wot, Dd, bo, attnf, nullptr, Dd, Dd);
  residual_ln_kernel<<<NR, 256, 0, stream>>>(attnf, x, g1, b1, x1f, x1bf);

  // 6) FFN: relu(x1 W1 + bw1) W2 + bw2 -> residual + LN2 -> out
  gemm_wmma_bf16<1, 1><<<gFfn1, 32, 0, stream>>>(x1bf, Dd, w1t, Dd, bw1, nullptr, hbf, FFNd, Dd);
  gemm_wmma_bf16<0, 0><<<gProj, 32, 0, stream>>>(hbf, FFNd, w2t, FFNd, bw2, ffn2f, nullptr, Dd, FFNd);
  residual_ln_kernel<<<NR, 256, 0, stream>>>(ffn2f, x1f, g2, b2, out, nullptr);
}
